// SpatialLocalSum_22308060135554
// MI455X (gfx1250) — matrix operand order repaired and checked by reading, not compiled, not run on hardware
//
#include <hip/hip_runtime.h>
#include <hip/hip_fp16.h>
#include <stdint.h>

typedef __attribute__((ext_vector_type(16))) _Float16 v16h;
typedef __attribute__((ext_vector_type(8)))  _Float16 h8;
typedef __attribute__((ext_vector_type(4)))  _Float16 h4;
typedef __attribute__((ext_vector_type(8)))  float    v8f;
typedef __attribute__((ext_vector_type(4)))  float    f4;

// Problem constants from the reference: B=128, H=32, W=32, C=64, S=64
#define LDA 72   // padded LDS stride (halves) for A  (128 x 64 tile)
#define LDB 72   // padded LDS stride (halves) for Bt (64 x 64 tile, transposed)

__global__ __launch_bounds__(256)
void slsum_wmma_kernel(const float* __restrict__ x,
                       const float* __restrict__ acc,
                       float* __restrict__ out)
{
    __shared__ __align__(16) float    acc_s[64 * 64];      // raw acc slice [c][s]
    __shared__ __align__(16) _Float16 A_s[128 * LDA];      // exp(x-mx) f16, [b][c]
    __shared__ __align__(16) _Float16 Bt_s[64 * LDB];      // acc/max_c  f16, [s][c]
    __shared__ float mw_s[64];
    __shared__ float inv_max_s[64];
    __shared__ float mx_s[128];

    const int t    = threadIdx.x;
    const int lane = t & 31;
    const int wave = t >> 5;
    const int hw   = blockIdx.x;                    // h*32 + w, 1024 blocks

    constexpr float LOG2E = 1.4426950408889634f;
    constexpr float LN2   = 0.6931471805599453f;

    // ---- Phase 1: async-stage acc[hw] slice (64x64 f32 = 16KB) straight into LDS.
    // GLOBAL_LOAD_ASYNC_TO_LDS_B128: no VGPR landing, tracked by ASYNCcnt; it
    // overlaps with all the x-row processing below.
    {
        const uint64_t sbase   = (uint64_t)(uintptr_t)(acc + (size_t)hw * 4096);
        const uint32_t ldsbase = (uint32_t)(uintptr_t)&acc_s[0];   // low 32 bits = LDS offset
        #pragma unroll
        for (int i = 0; i < 4; ++i) {
            uint32_t voff = (uint32_t)(i * 256 + t) * 16u;         // per-lane global byte offset
            uint32_t ldso = ldsbase + voff;                        // per-lane LDS byte offset
            asm volatile("global_load_async_to_lds_b128 %0, %1, %2"
                         :: "v"(ldso), "v"(voff), "s"(sbase)
                         : "memory");
        }
    }

    // ---- Phase 2: load x rows (2 threads per batch row), row-max via shfl, exp -> A ----
    {
        const int b    = t >> 1;
        const int half = t & 1;
        const f4* xrow = (const f4*)(x + ((size_t)b * 1024 + hw) * 64 + half * 32);
        f4 xv[8];
        #pragma unroll
        for (int i = 0; i < 8; ++i) xv[i] = xrow[i];

        float m = xv[0].x;
        #pragma unroll
        for (int i = 0; i < 8; ++i)
            m = fmaxf(m, fmaxf(fmaxf(xv[i].x, xv[i].y), fmaxf(xv[i].z, xv[i].w)));
        m = fmaxf(m, __shfl_xor(m, 1));             // combine the pair (wave32)
        if (half == 0) mx_s[b] = m;

        // exp(x-m) in (0,1]: raw v_exp_f32 is safe (no denorm-input concern)
        _Float16* arow = &A_s[b * LDA + half * 32];
        #pragma unroll
        for (int i = 0; i < 8; ++i) {
            h4 hv;
            hv[0] = (_Float16)__builtin_amdgcn_exp2f((xv[i].x - m) * LOG2E);
            hv[1] = (_Float16)__builtin_amdgcn_exp2f((xv[i].y - m) * LOG2E);
            hv[2] = (_Float16)__builtin_amdgcn_exp2f((xv[i].z - m) * LOG2E);
            hv[3] = (_Float16)__builtin_amdgcn_exp2f((xv[i].w - m) * LOG2E);
            *(h4*)(arow + i * 4) = hv;              // 8B aligned
        }
    }

    // Async LDS writes must complete before the barrier releases readers.
    asm volatile("s_wait_asynccnt 0x0" ::: "memory");
    __syncthreads();

    // ---- Phase 3: per-s max & sum over c (64 threads; conflict-free column reads) ----
    if (t < 64) {
        const int s = t;
        float mv = 0.0f, sv = 0.0f;                 // acc > 0
        #pragma unroll 4
        for (int c = 0; c < 64; ++c) {
            float v = acc_s[c * 64 + s];
            mv = fmaxf(mv, v);
            sv += v;
        }
        // mw = log(max) - log(sum); inputs in [1e-4, 64] — raw v_log_f32 safe
        mw_s[s]      = (__builtin_amdgcn_logf(mv) - __builtin_amdgcn_logf(sv)) * LN2;
        inv_max_s[s] = 1.0f / mv;
    }
    __syncthreads();

    // ---- Phase 4: Bt[s][c] = f16(acc[c][s] / max_c) (transposed for WMMA B frags) ----
    #pragma unroll
    for (int i = 0; i < 16; ++i) {
        int idx = i * 256 + t;                      // 0..4095, coalesced acc_s reads
        int c = idx >> 6, s = idx & 63;
        Bt_s[s * LDB + c] = (_Float16)(acc_s[idx] * inv_max_s[s]);
    }
    __syncthreads();

    // ---- Phase 5: WMMA. Wave w owns M rows [16w,16w+16), all 64 N, K=64 ----
    const int mrow = wave * 16 + (lane & 15);
    const int akb  = (lane >> 4) * 8;               // A K sub-base per lane half
    const int bkb  = (lane >> 4) * 8;               // B K sub-base per lane half
    const int scol = lane & 15;

    // Preload both A fragments (k0 = 0, 32); reused across all 4 N tiles.
    v16h afrag[2];
    #pragma unroll
    for (int kk = 0; kk < 2; ++kk) {
        const _Float16* ap = &A_s[mrow * LDA + kk * 32 + akb];
        h8 lo = *(const h8*)ap;                     // K = base .. base+7
        h8 hi = *(const h8*)(ap + 16);              // K = base+16 .. base+23
        #pragma unroll
        for (int j = 0; j < 8; ++j) { afrag[kk][j] = lo[j]; afrag[kk][8 + j] = hi[j]; }
    }

    #pragma unroll
    for (int nt = 0; nt < 4; ++nt) {
        v8f c = {};
        #pragma unroll
        for (int kk = 0; kk < 2; ++kk) {
            const _Float16* bp = &Bt_s[(nt * 16 + scol) * LDB + kk * 32 + bkb];
            h8 lo = *(const h8*)bp;                 // K = base .. base+7
            h8 hi = *(const h8*)(bp + 16);          // K = base+16 .. base+23
            v16h bfrag;
            #pragma unroll
            for (int j = 0; j < 8; ++j) { bfrag[j] = lo[j]; bfrag[8 + j] = hi[j]; }
            c = __builtin_amdgcn_wmma_f32_16x16x32_f16(
                    /*neg_a=*/false, afrag[kk], /*neg_b=*/false, bfrag,
                    /*c_mod=*/(short)0, c, /*reuse_a=*/false, /*reuse_b=*/false);
        }
        // Epilogue: out[b, hw, s] = log(prod) + mx[b] + mw[s]
        // prod >= W[cmax,s] >= ~1e-4 (f32-normal), so raw v_log_f32 is safe.
        const int   s   = nt * 16 + scol;
        const float mws = mw_s[s];
        #pragma unroll
        for (int r = 0; r < 8; ++r) {
            const int bb = wave * 16 + (lane >> 4) * 8 + r;   // C/D: VGPR r -> M row
            out[((size_t)bb * 1024 + hw) * 64 + s] =
                __builtin_amdgcn_logf(c[r]) * LN2 + mx_s[bb] + mws;
        }
    }
}

extern "C" void kernel_launch(void* const* d_in, const int* in_sizes, int n_in,
                              void* d_out, int out_size, void* d_ws, size_t ws_size,
                              hipStream_t stream) {
    (void)in_sizes; (void)n_in; (void)out_size; (void)d_ws; (void)ws_size;
    const float* x    = (const float*)d_in[0];   // [128, 32, 32, 64] f32
    const float* accw = (const float*)d_in[1];   // [32, 32, 64, 64] f32
    float*       out  = (float*)d_out;           // [128, 32, 32, 64] f32
    slsum_wmma_kernel<<<dim3(32 * 32), dim3(256), 0, stream>>>(x, accw, out);
}
